// VSSM_77197742178476
// MI455X (gfx1250) — matrix operand order repaired and checked by reading, not compiled, not run on hardware
//
#include <hip/hip_runtime.h>

// ---------------------------------------------------------------------------
// VMamba U-Net forward for gfx1250 (MI455X). GEMMs via v_wmma_f32_16x16x32_bf16
// (32x64 output strip per wave, 8 WMMAs / 12 b128-loads per k-step),
// selective scan via wave32 register recurrence + shfl reductions.
// ---------------------------------------------------------------------------

typedef unsigned short u16;
typedef __attribute__((ext_vector_type(16))) __bf16 v16bf;
typedef __attribute__((ext_vector_type(8)))  float  v8f;

#define DEVFN __device__ __forceinline__

DEVFN u16 f2bf(float f) {                       // fp32 -> bf16 RNE
  unsigned u = __float_as_uint(f);
  u += 0x7fffu + ((u >> 16) & 1u);
  return (u16)(u >> 16);
}
DEVFN float bf2f(u16 h) { return __uint_as_float(((unsigned)h) << 16); }
DEVFN float siluf(float x) { return x / (1.0f + __expf(-x)); }
DEVFN float softplusf(float x) { return x > 20.0f ? x : log1pf(__expf(x)); }

#define WMMA_BF16(A_, B_, C_) \
  __builtin_amdgcn_wmma_f32_16x16x32_bf16(false, (A_), false, (B_), (short)0, (C_), false, false)

// ---------------------------------------------------------------------------
// WMMA GEMM: C[M,N] = A[M,K] * B[N,K]^T   (A,B bf16 row-major, K % 32 == 0,
// lda/ldb % 8 == 0 so 128-bit loads are aligned).
// Each wave: 32(M) x 64(N) strip -> 8 accumulators, A reused 4x, B reused 2x.
// Block = 4 waves = 64(M) x 128(N) tile.
// ---------------------------------------------------------------------------
__global__ __launch_bounds__(128) void gemm_bf16_wmma(
    const u16* __restrict__ A, int lda,
    const u16* __restrict__ B, int ldb,
    float* __restrict__ C, int ldc, int M, int N, int K)
{
  const int lane = threadIdx.x & 31;
  const int wave = threadIdx.x >> 5;
  const int half = lane >> 4;
  const int l16  = lane & 15;
  const int tm = (blockIdx.y * 2 + (wave >> 1)) * 32;
  const int tn = (blockIdx.x * 2 + (wave & 1)) * 64;
  if (tm >= M || tn >= N) return;            // wave-uniform

  // A fragment rows (lane<16 -> K 0..15 contiguous; lane>=16 -> K 16..31)
  const u16* ap0 = A + (long)min(tm +      l16, M - 1) * lda + half * 16;
  const u16* ap1 = A + (long)min(tm + 16 + l16, M - 1) * lda + half * 16;
  // B fragment rows (lane<16 -> K 0..15 contiguous; lane>=16 -> K 16..31)
  const u16* bp0 = B + (long)min(tn +      l16, N - 1) * ldb + half * 16;
  const u16* bp1 = B + (long)min(tn + 16 + l16, N - 1) * ldb + half * 16;
  const u16* bp2 = B + (long)min(tn + 32 + l16, N - 1) * ldb + half * 16;
  const u16* bp3 = B + (long)min(tn + 48 + l16, N - 1) * ldb + half * 16;

  v8f acc[8] = {};
  union U { v16bf v; uint4 q[2]; };
  for (int k0 = 0; k0 < K; k0 += 32) {
    __builtin_prefetch(ap0 + k0 + 256, 0, 1);   // global_prefetch_b8, 8 iters ahead
    __builtin_prefetch(bp0 + k0 + 256, 0, 1);
    U a0, a1, b0, b1, b2, b3;
    a0.q[0] = *(const uint4*)(ap0 + k0); a0.q[1] = *(const uint4*)(ap0 + k0 + 8);
    a1.q[0] = *(const uint4*)(ap1 + k0); a1.q[1] = *(const uint4*)(ap1 + k0 + 8);
    b0.q[0] = *(const uint4*)(bp0 + k0); b0.q[1] = *(const uint4*)(bp0 + k0 + 8);
    b1.q[0] = *(const uint4*)(bp1 + k0); b1.q[1] = *(const uint4*)(bp1 + k0 + 8);
    b2.q[0] = *(const uint4*)(bp2 + k0); b2.q[1] = *(const uint4*)(bp2 + k0 + 8);
    b3.q[0] = *(const uint4*)(bp3 + k0); b3.q[1] = *(const uint4*)(bp3 + k0 + 8);
    acc[0] = WMMA_BF16(a0.v, b0.v, acc[0]);
    acc[1] = WMMA_BF16(a0.v, b1.v, acc[1]);
    acc[2] = WMMA_BF16(a0.v, b2.v, acc[2]);
    acc[3] = WMMA_BF16(a0.v, b3.v, acc[3]);
    acc[4] = WMMA_BF16(a1.v, b0.v, acc[4]);
    acc[5] = WMMA_BF16(a1.v, b1.v, acc[5]);
    acc[6] = WMMA_BF16(a1.v, b2.v, acc[6]);
    acc[7] = WMMA_BF16(a1.v, b3.v, acc[7]);
  }
  // C/D layout: VGPR i -> M = tile_m + i + half*8 ; N = tile_n + (lane&15)
#pragma unroll
  for (int mi = 0; mi < 2; ++mi)
#pragma unroll
    for (int nj = 0; nj < 4; ++nj) {
      v8f r = acc[mi * 4 + nj];
#pragma unroll
      for (int i = 0; i < 8; ++i) {
        int mm = tm + mi * 16 + i + half * 8;
        int nn = tn + nj * 16 + l16;
        if (mm < M && nn < N) C[(long)mm * ldc + nn] = r[i];
      }
    }
}

// ---------------------------------------------------------------------------
// fp32 -> bf16 with column zero-padding (K padding for WMMA)
// ---------------------------------------------------------------------------
__global__ void cvt_pad(const float* __restrict__ src, int ss,
                        u16* __restrict__ dst, int dld, int rows, int cols)
{
  long i = (long)blockIdx.x * blockDim.x + threadIdx.x;
  long total = (long)rows * dld;
  if (i >= total) return;
  int c = (int)(i % dld);
  long r = i / dld;
  dst[i] = (c < cols) ? f2bf(src[r * (long)ss + c]) : (u16)0;
}

// ---------------------------------------------------------------------------
// Elementwise / structural kernels
// ---------------------------------------------------------------------------
__global__ void patch_embed(const float* __restrict__ x, const float* __restrict__ w,
                            const float* __restrict__ b, float* __restrict__ out)
{ // x:[3,256,256]  w:[96,3,4,4]  out:[4096,96] (HWC)
  int idx = blockIdx.x * blockDim.x + threadIdx.x;
  if (idx >= 64 * 64 * 96) return;
  int c = idx % 96, l = idx / 96, ph = l / 64, pw = l % 64;
  float acc = b[c];
  for (int ci = 0; ci < 3; ++ci)
    for (int ky = 0; ky < 4; ++ky)
      for (int kx = 0; kx < 4; ++kx)
        acc += x[(ci * 256 + ph * 4 + ky) * 256 + pw * 4 + kx] *
               w[((c * 3 + ci) * 4 + ky) * 4 + kx];
  out[idx] = acc;
}

__global__ void layernorm(const float* __restrict__ src, float* __restrict__ dst,
                          const float* __restrict__ g, const float* __restrict__ b,
                          int cols)
{ // one wave per row (in-place safe)
  long row = blockIdx.x;
  const float* p = src + row * cols;
  float s = 0.f;
  for (int c = threadIdx.x; c < cols; c += 32) s += p[c];
  for (int o = 16; o; o >>= 1) s += __shfl_xor(s, o, 32);
  float mean = s / cols;
  float v = 0.f;
  for (int c = threadIdx.x; c < cols; c += 32) { float d = p[c] - mean; v += d * d; }
  for (int o = 16; o; o >>= 1) v += __shfl_xor(v, o, 32);
  float rstd = rsqrtf(v / cols + 1e-5f);
  float* q = dst + row * cols;
  for (int c = threadIdx.x; c < cols; c += 32) q[c] = (p[c] - mean) * rstd * g[c] + b[c];
}

__global__ void dwconv_silu(const float* __restrict__ xz, int ld,
                            const float* __restrict__ cw, const float* __restrict__ cb,
                            float* __restrict__ out, int H, int W, int Di)
{ // xc = first Di columns of xz[L, 2Di]; depthwise 3x3 SAME + bias + SiLU
  long idx = (long)blockIdx.x * blockDim.x + threadIdx.x;
  long total = (long)H * W * Di;
  if (idx >= total) return;
  int c = (int)(idx % Di);
  int l = (int)(idx / Di);
  int h = l / W, w = l % W;
  float acc = cb[c];
  for (int dy = -1; dy <= 1; ++dy)
    for (int dx = -1; dx <= 1; ++dx) {
      int hh = h + dy, ww = w + dx;
      if (hh >= 0 && hh < H && ww >= 0 && ww < W)
        acc += xz[(long)(hh * W + ww) * ld + c] * cw[c * 9 + (dy + 1) * 3 + (dx + 1)];
    }
  out[idx] = siluf(acc);
}

__global__ void gather_xs(const float* __restrict__ xc, u16* __restrict__ xs4,
                          int H, int W, int Di)
{ // build 4 scan orders (hw, wh, flipped) as bf16 [4, L, Di]
  int L = H * W;
  long idx = (long)blockIdx.x * blockDim.x + threadIdx.x;
  long total = (long)4 * L * Di;
  if (idx >= total) return;
  int c = (int)(idx % Di);
  long t = idx / Di;
  int l = (int)(t % L);
  int k = (int)(t / L);
  int src;
  if (k == 0)      src = l;
  else if (k == 1) src = (l % H) * W + (l / H);
  else if (k == 2) src = L - 1 - l;
  else { int p = L - 1 - l; src = (p % H) * W + (p / H); }
  xs4[idx] = f2bf(xc[(long)src * Di + c]);
}

__global__ void softplus_bias(float* __restrict__ dt, const float* __restrict__ dtb,
                              int L, int Di)
{
  long idx = (long)blockIdx.x * blockDim.x + threadIdx.x;
  long total = (long)4 * L * Di;
  if (idx >= total) return;
  int c = (int)(idx % Di);
  int k = (int)(idx / ((long)L * Di));
  dt[idx] = softplusf(dt[idx] + dtb[k * Di + c]);
}

__global__ void negexp(const float* __restrict__ s, float* __restrict__ d, int n)
{ int i = blockIdx.x * blockDim.x + threadIdx.x; if (i < n) d[i] = -__expf(s[i]); }

__global__ void zero_f32(float* p, int n)
{ int i = blockIdx.x * blockDim.x + threadIdx.x; if (i < n) p[i] = 0.f; }

__global__ void add_inplace(float* __restrict__ a, const float* __restrict__ b, int n)
{ int i = blockIdx.x * blockDim.x + threadIdx.x; if (i < n) a[i] += b[i]; }

__global__ void mul_silu(float* __restrict__ y, const float* __restrict__ xz,
                         int ld, int Di, int n)
{ // y[l,c] *= silu(z[l,c]) with z = xz[:, Di:2Di]
  int i = blockIdx.x * blockDim.x + threadIdx.x;
  if (i >= n) return;
  int c = i % Di;
  long r = i / Di;
  y[i] *= siluf(xz[r * ld + Di + c]);
}

// Selective scan: one wave handles 2 channels; lane = (group, state n of 16).
__global__ __launch_bounds__(32) void scan_dir(
    const float* __restrict__ delta,  // [L, Di]  this direction
    const u16*   __restrict__ xs,     // [L, Di]  this direction (bf16)
    const float* __restrict__ xdbl,   // [L, ldx] this direction (B @ R.., C @ R+16..)
    int ldx, int Roff,
    const float* __restrict__ Aneg,   // [Di, 16]
    const float* __restrict__ Ds,     // [Di]
    float* __restrict__ ysum,         // [L, Di]  hw order, accumulated
    int L, int Di, int H, int W, int kdir)
{
  int lane = threadIdx.x;
  int grp = lane >> 4, n = lane & 15;
  int c = blockIdx.x * 2 + grp;
  float a = Aneg[(long)c * 16 + n];
  float dcoef = Ds[c];
  float h = 0.f;
  for (int l = 0; l < L; ++l) {
    float dl = delta[(long)l * Di + c];
    float xv = bf2f(xs[(long)l * Di + c]);
    float Bv = xdbl[(long)l * ldx + Roff + n];
    float Cv = xdbl[(long)l * ldx + Roff + 16 + n];
    h = __expf(dl * a) * h + dl * xv * Bv;
    float yv = h * Cv;
    yv += __shfl_xor(yv, 1, 32);
    yv += __shfl_xor(yv, 2, 32);
    yv += __shfl_xor(yv, 4, 32);
    yv += __shfl_xor(yv, 8, 32);
    if (n == 0) {
      int lhw;
      if (kdir == 0)      lhw = l;
      else if (kdir == 1) lhw = (l % H) * W + (l / H);
      else if (kdir == 2) lhw = L - 1 - l;
      else { int p = L - 1 - l; lhw = (p % H) * W + (p / H); }
      ysum[(long)lhw * Di + c] += yv + dcoef * xv;   // unique (l,c) per launch
    }
  }
}

__global__ void merge_gather(const float* __restrict__ x, float* __restrict__ out,
                             int H2, int W2, int d)
{ // x:[2H2,2W2,d] -> out:[H2*W2, 4d], quadrant order (0,0)(1,0)(0,1)(1,1)
  long idx = (long)blockIdx.x * blockDim.x + threadIdx.x;
  long total = (long)H2 * W2 * 4 * d;
  if (idx >= total) return;
  int c4 = (int)(idx % (4 * d));
  long l = idx / (4 * d);
  int q = c4 / d, c = c4 % d;
  int h2 = (int)(l / W2), w2 = (int)(l % W2);
  int ry = q & 1, rx = q >> 1;
  out[idx] = x[((long)(2 * h2 + ry) * (2 * W2) + (2 * w2 + rx)) * d + c];
}

__global__ void expand_rearrange(const float* __restrict__ e, float* __restrict__ out,
                                 int H, int W, int ds, int co)
{ // e:[H*W, ds*ds*co] -> out:[(H*ds)*(W*ds), co]
  long idx = (long)blockIdx.x * blockDim.x + threadIdx.x;
  long total = (long)H * W * ds * ds * co;
  if (idx >= total) return;
  int c = (int)(idx % co);
  long l = idx / co;
  int Wo = W * ds;
  int ho = (int)(l / Wo), wo = (int)(l % Wo);
  int h = ho / ds, r = ho % ds, w = wo / ds, s = wo % ds;
  out[idx] = e[((long)(h * W + w)) * (ds * ds * co) + (r * ds + s) * co + c];
}

__global__ void classify(const float* __restrict__ hf, const float* __restrict__ cw,
                         const float* __restrict__ cb, float* __restrict__ out,
                         int L, int C, int O)
{
  int idx = blockIdx.x * blockDim.x + threadIdx.x;
  if (idx >= O * L) return;
  int o = idx / L, l = idx % L;
  float acc = cb[o];
  for (int c = 0; c < C; ++c) acc += hf[(long)l * C + c] * cw[o * C + c];
  out[(long)o * L + l] = acc;
}

// ---------------------------------------------------------------------------
// Host orchestration
// ---------------------------------------------------------------------------
struct Alloc { char* base; size_t off; };
static inline void* balloc(Alloc& a, size_t bytes) {
  size_t o = (a.off + 255) & ~(size_t)255;
  a.off = o + bytes;
  return a.base + o;
}
static inline float* fall(Alloc& a, size_t n) { return (float*)balloc(a, n * 4); }
static inline u16*   hall(Alloc& a, size_t n) { return (u16*)balloc(a, n * 2); }

struct BlockP {
  const float *A_logs, *Ds, *conv_b, *conv_w, *dt_b, *dt_w, *in_proj,
              *ln_b, *ln_g, *on_b, *on_g, *out_proj, *x_proj;
};
static void read_block(void* const* d_in, int& ii, BlockP& b) {
  b.A_logs  = (const float*)d_in[ii++]; b.Ds      = (const float*)d_in[ii++];
  b.conv_b  = (const float*)d_in[ii++]; b.conv_w  = (const float*)d_in[ii++];
  b.dt_b    = (const float*)d_in[ii++]; b.dt_w    = (const float*)d_in[ii++];
  b.in_proj = (const float*)d_in[ii++]; b.ln_b    = (const float*)d_in[ii++];
  b.ln_g    = (const float*)d_in[ii++]; b.on_b    = (const float*)d_in[ii++];
  b.on_g    = (const float*)d_in[ii++]; b.out_proj= (const float*)d_in[ii++];
  b.x_proj  = (const float*)d_in[ii++];
}

static void launch_gemm(hipStream_t st, const u16* A, int lda, const u16* B, int ldb,
                        float* C, int ldc, int M, int N, int K) {
  dim3 grid((N + 127) / 128, (M + 63) / 64);
  gemm_bf16_wmma<<<grid, 128, 0, st>>>(A, lda, B, ldb, C, ldc, M, N, K);
}
static void launch_cvt(hipStream_t st, const float* src, int ss, u16* dst, int dld,
                       int rows, int cols) {
  long n = (long)rows * dld;
  cvt_pad<<<(unsigned)((n + 255) / 256), 256, 0, st>>>(src, ss, dst, dld, rows, cols);
}
#define EL(n) <<<(unsigned)(((long)(n) + 255) / 256), 256, 0, st>>>

static void vss_block(hipStream_t st, const BlockP& P, float* h, int H, int W, int d,
                      Alloc al /* by value: auto-reset scratch */) {
  const int L = H * W, Di = 2 * d;
  const int R = (d + 15) / 16;
  const int Cp = R + 32;                 // R + 2N columns of x_dbl
  const int Rp = (R + 31) & ~31;         // K-padded R

  float* ln1 = fall(al, (size_t)L * d);
  layernorm<<<L, 32, 0, st>>>(h, ln1, P.ln_g, P.ln_b, d);

  u16* ln1b = hall(al, (size_t)L * d);
  launch_cvt(st, ln1, d, ln1b, d, L, d);
  u16* wip = hall(al, (size_t)2 * Di * d);
  launch_cvt(st, P.in_proj, d, wip, d, 2 * Di, d);

  float* xz = fall(al, (size_t)L * 2 * Di);
  launch_gemm(st, ln1b, d, wip, d, xz, 2 * Di, L, 2 * Di, d);

  float* xconv = fall(al, (size_t)L * Di);
  dwconv_silu EL((long)L * Di)(xz, 2 * Di, P.conv_w, P.conv_b, xconv, H, W, Di);

  u16* xs4b = hall(al, (size_t)4 * L * Di);
  gather_xs EL((long)4 * L * Di)(xconv, xs4b, H, W, Di);

  u16* xpb = hall(al, (size_t)4 * Cp * Di);
  launch_cvt(st, P.x_proj, Di, xpb, Di, 4 * Cp, Di);

  float* xdbl = fall(al, (size_t)4 * L * Cp);
  for (int k = 0; k < 4; ++k)
    launch_gemm(st, xs4b + (size_t)k * L * Di, Di, xpb + (size_t)k * Cp * Di, Di,
                xdbl + (size_t)k * L * Cp, Cp, L, Cp, Di);

  u16* dtsb = hall(al, (size_t)4 * L * Rp);
  for (int k = 0; k < 4; ++k)
    launch_cvt(st, xdbl + (size_t)k * L * Cp, Cp, dtsb + (size_t)k * L * Rp, Rp, L, R);
  u16* dtwb = hall(al, (size_t)4 * Di * Rp);
  for (int k = 0; k < 4; ++k)
    launch_cvt(st, P.dt_w + (size_t)k * Di * R, R, dtwb + (size_t)k * Di * Rp, Rp, Di, R);

  float* delta = fall(al, (size_t)4 * L * Di);
  for (int k = 0; k < 4; ++k)
    launch_gemm(st, dtsb + (size_t)k * L * Rp, Rp, dtwb + (size_t)k * Di * Rp, Rp,
                delta + (size_t)k * L * Di, Di, L, Di, Rp);
  softplus_bias EL((long)4 * L * Di)(delta, P.dt_b, L, Di);

  float* Aneg = fall(al, (size_t)4 * Di * 16);
  negexp EL(4 * Di * 16)(P.A_logs, Aneg, 4 * Di * 16);

  float* ysum = fall(al, (size_t)L * Di);
  zero_f32 EL(L * Di)(ysum, L * Di);
  for (int k = 0; k < 4; ++k)
    scan_dir<<<Di / 2, 32, 0, st>>>(delta + (size_t)k * L * Di, xs4b + (size_t)k * L * Di,
                                    xdbl + (size_t)k * L * Cp, Cp, R,
                                    Aneg + (size_t)k * Di * 16, P.Ds + (size_t)k * Di,
                                    ysum, L, Di, H, W, k);

  layernorm<<<L, 32, 0, st>>>(ysum, ysum, P.on_g, P.on_b, Di);
  mul_silu EL((long)L * Di)(ysum, xz, 2 * Di, Di, L * Di);

  u16* ynb = hall(al, (size_t)L * Di);
  launch_cvt(st, ysum, Di, ynb, Di, L, Di);
  u16* opb = hall(al, (size_t)d * Di);
  launch_cvt(st, P.out_proj, Di, opb, Di, d, Di);
  float* proj = fall(al, (size_t)L * d);
  launch_gemm(st, ynb, Di, opb, Di, proj, d, L, d, Di);
  add_inplace EL(L * d)(h, proj, L * d);
}

extern "C" void kernel_launch(void* const* d_in, const int* in_sizes, int n_in,
                              void* d_out, int out_size, void* d_ws, size_t ws_size,
                              hipStream_t stream) {
  (void)in_sizes; (void)n_in; (void)out_size; (void)ws_size;
  hipStream_t st = stream;
  int ii = 0;
  auto F = [&]() { return (const float*)d_in[ii++]; };

  // jax tree_leaves order: sorted dict keys at each level, lists in order.
  const float* x = F();
  const float* emb_b = F(); const float* emb_bln = F();
  const float* emb_g = F(); const float* emb_w  = F();
  const float* fin_b = F(); const float* fin_cb = F(); const float* fin_cw = F();
  const float* fin_exw = F(); const float* fin_g = F();

  BlockP enc[4][2]; const float *mg_b[3], *mg_g[3], *mg_w[3];
  for (int i = 0; i < 4; ++i) {
    read_block(d_in, ii, enc[i][0]); read_block(d_in, ii, enc[i][1]);
    if (i < 3) { mg_b[i] = F(); mg_g[i] = F(); mg_w[i] = F(); }
  }
  BlockP dec[4][2]; const float *ex_b[4] = {}, *ex_g[4] = {}, *ex_w[4] = {};
  for (int i = 0; i < 4; ++i) {
    read_block(d_in, ii, dec[i][0]); read_block(d_in, ii, dec[i][1]);
    if (i > 0) { ex_b[i] = F(); ex_g[i] = F(); ex_w[i] = F(); }
  }

  Alloc al{(char*)d_ws, 0};
  float* skipb[4] = {nullptr, fall(al, 1024 * 192), fall(al, 256 * 384), fall(al, 64 * 768)};
  float* hA = fall(al, 1572864);
  float* hB = fall(al, 1572864);
  float* h = hA; float* ho = hB;

  // ---- patch embed + LN ----
  patch_embed EL(4096 * 96)(x, emb_w, emb_b, h);
  layernorm<<<4096, 32, 0, st>>>(h, h, emb_g, emb_bln, 96);

  // ---- encoder ----
  for (int i = 0; i < 4; ++i) {
    int H = 64 >> i, d = 96 << i, L = H * H;
    if (i >= 1)
      hipMemcpyAsync(skipb[i], h, (size_t)L * d * 4, hipMemcpyDeviceToDevice, st);
    for (int bi = 0; bi < 2; ++bi) vss_block(st, enc[i][bi], h, H, H, d, al);
    if (i < 3) {
      int H2 = H / 2, L2 = H2 * H2, d4 = 4 * d, d2 = 2 * d;
      Alloc s = al;
      float* xc = fall(s, (size_t)L2 * d4);
      merge_gather EL((long)L2 * d4)(h, xc, H2, H2, d);
      layernorm<<<L2, 32, 0, st>>>(xc, xc, mg_g[i], mg_b[i], d4);
      u16* xcb = hall(s, (size_t)L2 * d4);
      launch_cvt(st, xc, d4, xcb, d4, L2, d4);
      u16* wb = hall(s, (size_t)d2 * d4);
      launch_cvt(st, mg_w[i], d4, wb, d4, d2, d4);
      launch_gemm(st, xcb, d4, wb, d4, ho, d2, L2, d2, d4);
      float* t = h; h = ho; ho = t;
    }
  }

  // ---- decoder ----
  for (int i = 0; i < 4; ++i) {
    int d = 768 >> i, H = 8 << i;
    if (i > 0) {
      int dp = 2 * d, Hp = H / 2, Lp = Hp * Hp;
      add_inplace EL(Lp * dp)(h, skipb[4 - i], Lp * dp);
      Alloc s = al;
      u16* hb = hall(s, (size_t)Lp * dp);
      launch_cvt(st, h, dp, hb, dp, Lp, dp);
      u16* wb = hall(s, (size_t)4 * d * dp);
      launch_cvt(st, ex_w[i], dp, wb, dp, 4 * d, dp);
      float* e = fall(s, (size_t)Lp * 4 * d);
      launch_gemm(st, hb, dp, wb, dp, e, 4 * d, Lp, 4 * d, dp);
      expand_rearrange EL((long)H * H * d)(e, ho, Hp, Hp, 2, d);
      layernorm<<<H * H, 32, 0, st>>>(ho, ho, ex_g[i], ex_b[i], d);
      float* t = h; h = ho; ho = t;
    }
    for (int bi = 0; bi < 2; ++bi) vss_block(st, dec[i][bi], h, H, H, d, al);
  }

  // ---- final expand (ds=4) + classifier ----
  {
    Alloc s = al;
    u16* hb = hall(s, (size_t)4096 * 96);
    launch_cvt(st, h, 96, hb, 96, 4096, 96);
    u16* wb = hall(s, (size_t)384 * 96);
    launch_cvt(st, fin_exw, 96, wb, 96, 384, 96);
    float* e = fall(s, (size_t)4096 * 384);
    launch_gemm(st, hb, 96, wb, 96, e, 384, 4096, 384, 96);
    float* hf = ho;                               // [65536, 24]
    expand_rearrange EL((long)65536 * 24)(e, hf, 64, 64, 4, 24);
    layernorm<<<65536, 32, 0, st>>>(hf, hf, fin_g, fin_b, 24);
    classify EL(9 * 65536)(hf, fin_cw, fin_cb, (float*)d_out, 65536, 24, 9);
  }
}
